// CorrectionHead_14645838479428
// MI455X (gfx1250) — compile-verified, matrix operand order checked by reading, not run on hardware
//
#include <hip/hip_runtime.h>
#include <hip/hip_bf16.h>
#include <math.h>

// ---------------------------------------------------------------------------
// Problem dims (fixed by the reference)
// ---------------------------------------------------------------------------
#define HDIM   2048      // H
#define IN3    6144      // 3*H  (GEMM1 K)
#define CHDIM  512       // CH   (GEMM1 N, GEMM2 K)
#define MROWS  16384     // B*S

typedef __attribute__((ext_vector_type(16))) __bf16 v16bf;
typedef __attribute__((ext_vector_type(8)))  float  v8f;

__device__ __forceinline__ unsigned short f32_to_bf16(float f) {
    union { float f; unsigned int u; } v; v.f = f;
    unsigned int r = v.u + 0x7FFFu + ((v.u >> 16) & 1u);   // round to nearest even
    return (unsigned short)(r >> 16);
}

// bf16 fragment: 16 elements per lane = two 16-byte global loads
union FragBF {
    uint4 q[2];
    v16bf v;
};

// Load a 16x32 bf16 A/B fragment for v_wmma_f32_16x16x32_bf16.
// ISA layout (CDNA5 ISA, 16-bit A 16x32): lane = (row&15) + 16*h,
//   h=0: elems 0..7 -> K = k0+0..7,  elems 8..15 -> K = k0+16..23
//   h=1: elems 0..7 -> K = k0+8..15, elems 8..15 -> K = k0+24..31
__device__ __forceinline__ void load_frag(const unsigned short* __restrict__ base,
                                          int ld, int row, int k0, int h, FragBF& f) {
    const unsigned short* p = base + (size_t)row * ld + k0 + h * 8;
    f.q[0] = *reinterpret_cast<const uint4*>(p);
    f.q[1] = *reinterpret_cast<const uint4*>(p + 16);
}

// ---------------------------------------------------------------------------
// Kernel 1: per-row LayerNorm of {prev, u_t, z_t} -> bf16 concat buffer
// grid = (MROWS, 3), block = 256 (8 waves); each thread owns 8 columns
// ---------------------------------------------------------------------------
__global__ __launch_bounds__(256)
void ln_concat_kernel(const float* __restrict__ u_t,
                      const float* __restrict__ z_t,
                      const float* __restrict__ prev,
                      const float* __restrict__ prev_g, const float* __restrict__ prev_b,
                      const float* __restrict__ u_g,    const float* __restrict__ u_b,
                      const float* __restrict__ z_g,    const float* __restrict__ z_b,
                      unsigned short* __restrict__ Xc)   // [MROWS, IN3] bf16
{
    const int row    = blockIdx.x;
    const int branch = blockIdx.y;                 // 0: prev, 1: u_t, 2: z_t
    const float* src; const float* g; const float* b;
    if (branch == 0)      { src = prev; g = prev_g; b = prev_b; }
    else if (branch == 1) { src = u_t;  g = u_g;    b = u_b;    }
    else                  { src = z_t;  g = z_g;    b = z_b;    }

    const int t = threadIdx.x;
    const float4* p = reinterpret_cast<const float4*>(src + (size_t)row * HDIM) + t * 2;
    float4 v0 = p[0];
    float4 v1 = p[1];
    float x[8] = { v0.x, v0.y, v0.z, v0.w, v1.x, v1.y, v1.z, v1.w };

    float s = 0.f, ss = 0.f;
    #pragma unroll
    for (int j = 0; j < 8; ++j) { s += x[j]; ss += x[j] * x[j]; }

    // wave32 butterfly reduce
    #pragma unroll
    for (int off = 16; off > 0; off >>= 1) {
        s  += __shfl_xor(s,  off, 32);
        ss += __shfl_xor(ss, off, 32);
    }
    __shared__ float sh_s[8], sh_ss[8];
    const int wave = t >> 5, lane = t & 31;
    if (lane == 0) { sh_s[wave] = s; sh_ss[wave] = ss; }
    __syncthreads();
    s = 0.f; ss = 0.f;
    #pragma unroll
    for (int w = 0; w < 8; ++w) { s += sh_s[w]; ss += sh_ss[w]; }

    const float mu   = s * (1.0f / HDIM);
    const float var  = ss * (1.0f / HDIM) - mu * mu;
    const float rstd = rsqrtf(var + 1e-5f);

    alignas(16) unsigned short outv[8];
    const int col0 = t * 8;
    #pragma unroll
    for (int j = 0; j < 8; ++j) {
        const int c = col0 + j;
        const float y = (x[j] - mu) * rstd * g[c] + b[c];
        outv[j] = f32_to_bf16(y);
    }
    uint4* dst = reinterpret_cast<uint4*>(Xc + (size_t)row * IN3 + branch * HDIM + col0);
    *dst = *reinterpret_cast<const uint4*>(outv);
}

// ---------------------------------------------------------------------------
// Kernel 2: elementwise f32 -> bf16 (weight conversion; weights are L2-hot)
// ---------------------------------------------------------------------------
__global__ __launch_bounds__(256)
void cvt_bf16_kernel(const float* __restrict__ src, unsigned short* __restrict__ dst, int n) {
    const int i = blockIdx.x * 256 + threadIdx.x;
    if (i < n) dst[i] = f32_to_bf16(src[i]);
}

// ---------------------------------------------------------------------------
// GEMM1: hidden = GELU( X[M,IN3] * W1[CH,IN3]^T + b1 ) -> bf16 [M, CH]
// block tile 64(M) x 256(N); 8 waves, each wave a 32x64 tile (2x4 WMMA accs).
// A (activations) is streamed with near-scope global_prefetch two K-tiles
// ahead; B (weights) is L2-resident.
// ---------------------------------------------------------------------------
__global__ __launch_bounds__(256)
void gemm1_gelu_kernel(const unsigned short* __restrict__ A,   // [MROWS, IN3] bf16
                       const unsigned short* __restrict__ B,   // [CHDIM, IN3] bf16
                       const float* __restrict__ bias,         // [CHDIM]
                       unsigned short* __restrict__ Hb)        // [MROWS, CHDIM] bf16
{
    const int lane  = threadIdx.x & 31;
    const int wave  = threadIdx.x >> 5;
    const int mWave = wave & 1;            // 2 waves along M
    const int nWave = wave >> 1;           // 4 waves along N
    const int mBase = blockIdx.x * 64  + mWave * 32;
    const int nBase = blockIdx.y * 256 + nWave * 64;
    const int r = lane & 15;
    const int h = lane >> 4;

    const unsigned short* aRow0 = A + (size_t)(mBase      + r) * IN3;
    const unsigned short* aRow1 = A + (size_t)(mBase + 16 + r) * IN3;

    v8f acc[2][4] = {};

    for (int k0 = 0; k0 < IN3; k0 += 32) {
        FragBF a[2], b[4];
        load_frag(A, IN3, mBase      + r, k0, h, a[0]);
        load_frag(A, IN3, mBase + 16 + r, k0, h, a[1]);
        #pragma unroll
        for (int ni = 0; ni < 4; ++ni)
            load_frag(B, IN3, nBase + ni * 16 + r, k0, h, b[ni]);

        // Pull the next-next A K-tile into the near (WGP) cache while the
        // WMMAs drain. locality=3 -> nearest-scope prefetch.
        if (k0 + 64 < IN3) {
            __builtin_prefetch(aRow0 + k0 + 64, 0, 3);
            __builtin_prefetch(aRow1 + k0 + 64, 0, 3);
        }

        #pragma unroll
        for (int mi = 0; mi < 2; ++mi)
            #pragma unroll
            for (int ni = 0; ni < 4; ++ni)
                acc[mi][ni] = __builtin_amdgcn_wmma_f32_16x16x32_bf16(
                    false, a[mi].v, false, b[ni].v, (short)0, acc[mi][ni], false, false);
    }

    // Epilogue: C/D layout -> n = nBase + ni*16 + r ; m = mBase + mi*16 + h*8 + j
    #pragma unroll
    for (int mi = 0; mi < 2; ++mi) {
        #pragma unroll
        for (int ni = 0; ni < 4; ++ni) {
            const int n  = nBase + ni * 16 + r;
            const float bn = bias[n];
            #pragma unroll
            for (int j = 0; j < 8; ++j) {
                const int m = mBase + mi * 16 + h * 8 + j;
                const float x = acc[mi][ni][j] + bn;
                const float y = 0.5f * x * (1.0f + erff(x * 0.70710678118654752f)); // exact GELU
                Hb[(size_t)m * CHDIM + n] = f32_to_bf16(y);
            }
        }
    }
}

// ---------------------------------------------------------------------------
// GEMM2: out = hidden[M,CH] * W2[H,CH]^T + b2 -> f32 [M, H]
// Same 64x256 block / 32x64 wave tiling; K = 512.
// ---------------------------------------------------------------------------
__global__ __launch_bounds__(256)
void gemm2_kernel(const unsigned short* __restrict__ A,   // [MROWS, CHDIM] bf16
                  const unsigned short* __restrict__ B,   // [HDIM, CHDIM] bf16
                  const float* __restrict__ bias,         // [HDIM]
                  float* __restrict__ Out)                 // [MROWS, HDIM] f32
{
    const int lane  = threadIdx.x & 31;
    const int wave  = threadIdx.x >> 5;
    const int mWave = wave & 1;
    const int nWave = wave >> 1;
    const int mBase = blockIdx.x * 64  + mWave * 32;
    const int nBase = blockIdx.y * 256 + nWave * 64;
    const int r = lane & 15;
    const int h = lane >> 4;

    const unsigned short* aRow0 = A + (size_t)(mBase      + r) * CHDIM;
    const unsigned short* aRow1 = A + (size_t)(mBase + 16 + r) * CHDIM;

    v8f acc[2][4] = {};

    for (int k0 = 0; k0 < CHDIM; k0 += 32) {
        FragBF a[2], b[4];
        load_frag(A, CHDIM, mBase      + r, k0, h, a[0]);
        load_frag(A, CHDIM, mBase + 16 + r, k0, h, a[1]);
        #pragma unroll
        for (int ni = 0; ni < 4; ++ni)
            load_frag(B, CHDIM, nBase + ni * 16 + r, k0, h, b[ni]);

        if (k0 + 64 < CHDIM) {
            __builtin_prefetch(aRow0 + k0 + 64, 0, 3);
            __builtin_prefetch(aRow1 + k0 + 64, 0, 3);
        }

        #pragma unroll
        for (int mi = 0; mi < 2; ++mi)
            #pragma unroll
            for (int ni = 0; ni < 4; ++ni)
                acc[mi][ni] = __builtin_amdgcn_wmma_f32_16x16x32_bf16(
                    false, a[mi].v, false, b[ni].v, (short)0, acc[mi][ni], false, false);
    }

    #pragma unroll
    for (int mi = 0; mi < 2; ++mi) {
        #pragma unroll
        for (int ni = 0; ni < 4; ++ni) {
            const int n  = nBase + ni * 16 + r;
            const float bn = bias[n];
            #pragma unroll
            for (int j = 0; j < 8; ++j) {
                const int m = mBase + mi * 16 + h * 8 + j;
                Out[(size_t)m * HDIM + n] = acc[mi][ni][j] + bn;
            }
        }
    }
}

// ---------------------------------------------------------------------------
// Launcher
// d_in order: 0 u_t, 1 z_t, 2 prev, 3 prev_g, 4 prev_b, 5 u_g, 6 u_b,
//             7 z_g, 8 z_b, 9 W1, 10 b1, 11 W2, 12 b2
// ---------------------------------------------------------------------------
extern "C" void kernel_launch(void* const* d_in, const int* in_sizes, int n_in,
                              void* d_out, int out_size, void* d_ws, size_t ws_size,
                              hipStream_t stream) {
    (void)in_sizes; (void)n_in; (void)out_size; (void)ws_size;

    const float* u_t    = (const float*)d_in[0];
    const float* z_t    = (const float*)d_in[1];
    const float* prev   = (const float*)d_in[2];
    const float* prev_g = (const float*)d_in[3];
    const float* prev_b = (const float*)d_in[4];
    const float* u_g    = (const float*)d_in[5];
    const float* u_b    = (const float*)d_in[6];
    const float* z_g    = (const float*)d_in[7];
    const float* z_b    = (const float*)d_in[8];
    const float* W1     = (const float*)d_in[9];
    const float* b1     = (const float*)d_in[10];
    const float* W2     = (const float*)d_in[11];
    const float* b2     = (const float*)d_in[12];
    float* out          = (float*)d_out;

    // Workspace carve (bf16 staging): Xc 201.3MB + W1b 6.3MB + W2b 2.1MB + Hb 16.8MB
    char* ws = (char*)d_ws;
    unsigned short* Xc  = (unsigned short*)(ws);
    unsigned short* W1b = (unsigned short*)(ws + (size_t)MROWS * IN3 * 2);
    unsigned short* W2b = (unsigned short*)(ws + (size_t)MROWS * IN3 * 2
                                               + (size_t)CHDIM * IN3 * 2);
    unsigned short* Hb  = (unsigned short*)(ws + (size_t)MROWS * IN3 * 2
                                               + (size_t)CHDIM * IN3 * 2
                                               + (size_t)HDIM * CHDIM * 2);

    // Weight conversion (small; W1b/W2b stay L2-resident afterwards)
    {
        const int n1 = CHDIM * IN3;     // 3,145,728
        const int n2 = HDIM * CHDIM;    // 1,048,576
        cvt_bf16_kernel<<<(n1 + 255) / 256, 256, 0, stream>>>(W1, W1b, n1);
        cvt_bf16_kernel<<<(n2 + 255) / 256, 256, 0, stream>>>(W2, W2b, n2);
    }

    // Fused 3x LayerNorm -> bf16 concat
    {
        dim3 grid(MROWS, 3);
        ln_concat_kernel<<<grid, 256, 0, stream>>>(u_t, z_t, prev,
                                                   prev_g, prev_b, u_g, u_b, z_g, z_b, Xc);
    }

    // GEMM1 + bias + exact GELU -> bf16 hidden
    {
        dim3 grid(MROWS / 64, CHDIM / 256);   // 256 x 2
        gemm1_gelu_kernel<<<grid, 256, 0, stream>>>(Xc, W1b, b1, Hb);
    }

    // GEMM2 + bias -> f32 output
    {
        dim3 grid(MROWS / 64, HDIM / 256);    // 256 x 8
        gemm2_kernel<<<grid, 256, 0, stream>>>(Hb, W2b, b2, out);
    }
}
